// Encoder_73985106640937
// MI455X (gfx1250) — compile-verified, matrix operand order checked by reading, not run on hardware
//
#include <hip/hip_runtime.h>
#include <hip/hip_bf16.h>
#include <stdint.h>

typedef __attribute__((ext_vector_type(16))) _Float16 v16h;
typedef __attribute__((ext_vector_type(8)))  float    v8f;

#define NB   2
#define NM   32
#define NT   512
#define ND   128
#define NH   4
#define NDK  32
#define NFF  512
#define NROW (NB * NM * NT)   // 32768 rows of D=128

__device__ __forceinline__ v8f wmma_f16(v16h a, v16h b, v8f c) {
  // D = A(16x32 f16) x B(32x16 f16) + C(16x16 f32)
  return __builtin_amdgcn_wmma_f32_16x16x32_f16(false, a, false, b, (short)0, c, false, false);
}

// A-operand (16 rows x 32 K, f16, row-major source, row stride ld).
// Also serves as B-operand when computing A x X^T (B columns == X rows).
// Layout per ISA: lane<16 -> row=lane, K 0-7 & 16-23; lane>=16 -> K 8-15 & 24-31.
__device__ __forceinline__ v16h load_rm(const _Float16* __restrict__ p, int ld) {
  int lane = threadIdx.x & 31;
  const _Float16* r = p + (lane & 15) * ld + ((lane >> 4) << 3);
  v16h o;
#pragma unroll
  for (int i = 0; i < 8; ++i) o[i] = r[i];
#pragma unroll
  for (int i = 0; i < 8; ++i) o[8 + i] = r[16 + i];
  return o;
}

// B-operand for C = A x Bmat, Bmat (32 x >=16) row-major: B[k][n] = p[k*ld+n].
// Used only on small LDS tiles (attn_m); bulk paths use transposed copies instead.
__device__ __forceinline__ v16h load_cm(const _Float16* __restrict__ p, int ld) {
  int lane = threadIdx.x & 31;
  int col  = lane & 15;
  int ko   = (lane >> 4) << 3;
  v16h o;
#pragma unroll
  for (int i = 0; i < 8; ++i) o[i] = p[(ko + i) * ld + col];
#pragma unroll
  for (int i = 0; i < 8; ++i) o[8 + i] = p[(16 + ko + i) * ld + col];
  return o;
}

// B-operand from a column-major (transposed) source: column n contiguous over k.
// ct[n][k] with column stride ldT: B[k][n] = ct[n*ldT + k].
__device__ __forceinline__ v16h load_tp(const _Float16* __restrict__ ct, int ldT) {
  int lane = threadIdx.x & 31;
  const _Float16* r = ct + (lane & 15) * ldT + ((lane >> 4) << 3);
  v16h o;
#pragma unroll
  for (int i = 0; i < 8; ++i) o[i] = r[i];
#pragma unroll
  for (int i = 0; i < 8; ++i) o[8 + i] = r[16 + i];
  return o;
}

__device__ __forceinline__ float rmax16(float x) {
#pragma unroll
  for (int d = 1; d < 16; d <<= 1) x = fmaxf(x, __shfl_xor(x, d, 32));
  return x;
}
__device__ __forceinline__ float rsum16(float x) {
#pragma unroll
  for (int d = 1; d < 16; d <<= 1) x += __shfl_xor(x, d, 32);
  return x;
}

// ---------------- LayerNorm (one wave per 128-float row) ----------------
template<bool F16OUT>
__global__ __launch_bounds__(256) void ln_kernel(const float* __restrict__ x,
                                                 const float* __restrict__ g,
                                                 const float* __restrict__ be,
                                                 _Float16* __restrict__ oh,
                                                 float* __restrict__ of) {
  int row  = (blockIdx.x * blockDim.x + threadIdx.x) >> 5;
  int lane = threadIdx.x & 31;
  const float* xr = x + (size_t)row * ND;
  float4 v = *(const float4*)(xr + lane * 4);
  float s = v.x + v.y + v.z + v.w;
#pragma unroll
  for (int d = 1; d < 32; d <<= 1) s += __shfl_xor(s, d, 32);
  float mu = s * (1.0f / ND);
  float dx = v.x - mu, dy = v.y - mu, dz = v.z - mu, dw = v.w - mu;
  float ss = dx * dx + dy * dy + dz * dz + dw * dw;
#pragma unroll
  for (int d = 1; d < 32; d <<= 1) ss += __shfl_xor(ss, d, 32);
  float inv = rsqrtf(ss * (1.0f / ND) + 1e-6f);
  float4 gg = *(const float4*)(g + lane * 4);
  float4 bb = *(const float4*)(be + lane * 4);
  float r0 = dx * inv * gg.x + bb.x;
  float r1 = dy * inv * gg.y + bb.y;
  float r2 = dz * inv * gg.z + bb.z;
  float r3 = dw * inv * gg.w + bb.w;
  size_t base = (size_t)row * ND + lane * 4;
  if (F16OUT) {
    oh[base + 0] = (_Float16)r0; oh[base + 1] = (_Float16)r1;
    oh[base + 2] = (_Float16)r2; oh[base + 3] = (_Float16)r3;
  } else {
    of[base + 0] = r0; of[base + 1] = r1;
    of[base + 2] = r2; of[base + 3] = r3;
  }
}

// ---- GEMM: out = A(f16, N x DI) x W(f16, DO x DI)^T + bias, 16x64 tile per wave ----
// F16OUT: row-major f16 out.  XPOSE: also/only write transposed f16 [bm][d][t] (DO==128).
// RESID (implies fp32 out): outF = resid + val.
template<int DI, bool RELU, bool F16OUT, bool RESID, bool XPOSE>
__global__ __launch_bounds__(256) void gemm_kernel(const _Float16* __restrict__ A,
                                                   const _Float16* __restrict__ W,
                                                   const float* __restrict__ bias,
                                                   _Float16* __restrict__ outH,
                                                   _Float16* __restrict__ outT,
                                                   float* __restrict__ outF,
                                                   const float* __restrict__ resid,
                                                   int DO) {
  int wave = (blockIdx.x * blockDim.x + threadIdx.x) >> 5;
  int tpr  = DO >> 6;                 // 64-wide column strips
  int rt   = wave / tpr;
  int cs   = wave - rt * tpr;
  const _Float16* a0 = A + (size_t)rt * 16 * DI;
  const _Float16* w0 = W + (size_t)cs * 64 * DI;
  v8f c0 = {}, c1 = {}, c2 = {}, c3 = {};
#pragma unroll
  for (int k = 0; k < DI; k += 32) {
    v16h a = load_rm(a0 + k, DI);
    c0 = wmma_f16(a, load_rm(w0 + (size_t)0 * 16 * DI + k, DI), c0);
    c1 = wmma_f16(a, load_rm(w0 + (size_t)1 * 16 * DI + k, DI), c1);
    c2 = wmma_f16(a, load_rm(w0 + (size_t)2 * 16 * DI + k, DI), c2);
    c3 = wmma_f16(a, load_rm(w0 + (size_t)3 * 16 * DI + k, DI), c3);
  }
  int lane = threadIdx.x & 31;
  int col  = lane & 15;
  int rb   = (lane >> 4) << 3;
  v8f acc[4] = {c0, c1, c2, c3};
#pragma unroll
  for (int n = 0; n < 4; ++n) {
    int colg = cs * 64 + n * 16 + col;
    float bv = bias[colg];
#pragma unroll
    for (int v = 0; v < 8; ++v) {
      float val = acc[n][v] + bv;
      if (RELU) val = fmaxf(val, 0.0f);
      int row = rt * 16 + rb + v;
      if (F16OUT) outH[(size_t)row * DO + colg] = (_Float16)val;
      if (XPOSE) {
        int t  = row & (NT - 1);
        int bm = row >> 9;            // NT == 512
        outT[((size_t)bm * ND + colg) * NT + t] = (_Float16)val;
      }
      if (!F16OUT && !XPOSE)
        outF[(size_t)row * DO + colg] = (RESID ? resid[(size_t)row * DO + colg] : 0.0f) + val;
    }
  }
}

// ---------------- Windowed (L=16) context attention over d=128 ----------------
__global__ __launch_bounds__(256) void ctx_attn_kernel(const _Float16* __restrict__ src,
                                                       const _Float16* __restrict__ srcT,
                                                       _Float16* __restrict__ dst) {
  __shared__ _Float16 Pl[8][16][32];
  int wave = (blockIdx.x * blockDim.x + threadIdx.x) >> 5;
  int qb   = wave & (NT / 16 - 1);
  int bm   = wave >> 5;
  const _Float16* seq = src + (size_t)bm * NT * ND;
  const _Float16* qp  = seq + (size_t)qb * 16 * ND;
  const _Float16* pp  = qb ? (qp - 16 * ND) : qp;   // prev block (dummy when qb==0)
  v8f s0 = {}, s1 = {};
#pragma unroll
  for (int k = 0; k < ND; k += 32) {
    v16h a  = load_rm(qp + k, ND);
    v16h bp = load_rm(pp + k, ND);
    s0 = wmma_f16(a, bp, s0);    // Q x Prev^T
    s1 = wmma_f16(a, a,  s1);    // Q x Cur^T
  }
  int lane = threadIdx.x & 31;
  int col  = lane & 15;
  int hf   = lane >> 4;
  const float scale = 0.088388347648318447f;   // 1/sqrt(128)
  _Float16 (*P)[32] = Pl[threadIdx.x >> 5];
#pragma unroll
  for (int v = 0; v < 8; ++v) {
    int row = v + (hf << 3);
    // band: prev key j valid iff j > row; cur key j valid iff j <= row.
    // zero padding (qb==0): in-band scores are 0 and DO join the softmax.
    float sp = (col > row)  ? (qb ? s0[v] * scale : 0.0f) : -1e30f;
    float sc = (col <= row) ? s1[v] * scale : -1e30f;
    float mx = rmax16(fmaxf(sp, sc));
    float ep = __expf(sp - mx);
    float ec = __expf(sc - mx);
    float invd = 1.0f / rsum16(ep + ec);
    float wp = qb ? ep * invd : 0.0f;  // zero rows: no contribution to numerator
    float wc = ec * invd;
    P[row][col]      = (_Float16)wp;
    P[row][16 + col] = (_Float16)wc;
  }
  __syncthreads();
  v16h pA = load_rm(&P[0][0], 32);
  const _Float16* sT = srcT + (size_t)bm * ND * NT;
  int tp0 = (qb ? qb - 1 : qb) * 16;
  int tc0 = qb * 16;
  int ko  = hf << 3;
  _Float16* dr = dst + (size_t)bm * NT * ND + (size_t)qb * 16 * ND;
#pragma unroll
  for (int n0 = 0; n0 < ND; n0 += 16) {
    const _Float16* colp = sT + (size_t)(n0 + col) * NT;  // contiguous over t
    v16h b;
#pragma unroll
    for (int i = 0; i < 8; ++i) b[i]     = colp[tp0 + ko + i];  // k 0..15  = prev rows
#pragma unroll
    for (int i = 0; i < 8; ++i) b[8 + i] = colp[tc0 + ko + i];  // k 16..31 = cur rows
    v8f o = {};
    o = wmma_f16(pA, b, o);
#pragma unroll
    for (int v = 0; v < 8; ++v)
      dr[(size_t)((hf << 3) + v) * ND + n0 + col] = (_Float16)o[v];
  }
}

// ---------------- Full attention over t=512 (flash-style), per (b,m,h,qblock) ----------------
__global__ __launch_bounds__(256) void attn_t_kernel(const _Float16* __restrict__ cq,
                                                     const _Float16* __restrict__ ck,
                                                     const _Float16* __restrict__ vT,
                                                     _Float16* __restrict__ out) {
  __shared__ _Float16 Pl[8][16][32];
  int wave = (blockIdx.x * blockDim.x + threadIdx.x) >> 5;
  int qb = wave & 31;
  int h  = (wave >> 5) & 3;
  int bm = wave >> 7;
  int b  = bm >> 5;
  int m  = bm & 31;
  size_t seqoff = (size_t)bm * NT * ND + h * NDK;
  const _Float16* Q  = cq + seqoff + (size_t)qb * 16 * ND;
  const _Float16* K  = ck + seqoff;
  const _Float16* VT = vT + (size_t)bm * ND * NT + (size_t)(h * NDK) * NT;  // [dk][t]
  int lane = threadIdx.x & 31;
  int col  = lane & 15;
  int hf   = lane >> 4;
  int ko   = hf << 3;
  _Float16 (*P)[32] = Pl[threadIdx.x >> 5];
  const float scale = 0.17677669529663689f;   // 1/sqrt(32)
  v16h qa = load_rm(Q, ND);
  v8f o0 = {}, o1 = {};
  float mrow[8], lrow[8];
#pragma unroll
  for (int v = 0; v < 8; ++v) { mrow[v] = -1e30f; lrow[v] = 0.0f; }
  for (int kb = 0; kb < NT / 32; ++kb) {
    const _Float16* Kb = K + (size_t)kb * 32 * ND;
    if (kb + 1 < NT / 32) {   // speculative prefetch of next key block -> global_prefetch_b8
      __builtin_prefetch(Kb + 32 * ND + lane * 64, 0, 0);
      __builtin_prefetch(VT + (size_t)col * NT + (kb + 1) * 32, 0, 0);
    }
    v8f s0 = {}, s1 = {};
    s0 = wmma_f16(qa, load_rm(Kb, ND), s0);
    s1 = wmma_f16(qa, load_rm(Kb + 16 * ND, ND), s1);
#pragma unroll
    for (int v = 0; v < 8; ++v) {
      float x0 = s0[v] * scale, x1 = s1[v] * scale;
      float mx = rmax16(fmaxf(x0, x1));
      float mn = fmaxf(mrow[v], mx);
      float corr = __expf(mrow[v] - mn);
      float p0 = __expf(x0 - mn);
      float p1 = __expf(x1 - mn);
      lrow[v] = lrow[v] * corr + rsum16(p0 + p1);
      mrow[v] = mn;
      o0[v] *= corr; o1[v] *= corr;
      int row = v + (hf << 3);
      P[row][col]      = (_Float16)p0;
      P[row][16 + col] = (_Float16)p1;
    }
    __syncthreads();
    v16h pA = load_rm(&P[0][0], 32);
    // B operand for P x V from transposed V: column dk contiguous over t
    const _Float16* vc0 = VT + (size_t)col * NT + kb * 32;
    const _Float16* vc1 = VT + (size_t)(16 + col) * NT + kb * 32;
    v16h bv0, bv1;
#pragma unroll
    for (int i = 0; i < 8; ++i) { bv0[i] = vc0[ko + i]; bv0[8 + i] = vc0[16 + ko + i]; }
#pragma unroll
    for (int i = 0; i < 8; ++i) { bv1[i] = vc1[ko + i]; bv1[8 + i] = vc1[16 + ko + i]; }
    o0 = wmma_f16(pA, bv0, o0);
    o1 = wmma_f16(pA, bv1, o1);
    __syncthreads();
  }
  // out layout: [b][t][h][m][dk]
#pragma unroll
  for (int v = 0; v < 8; ++v) {
    float inv = 1.0f / lrow[v];
    int t = qb * 16 + (hf << 3) + v;
    size_t base = ((((size_t)b * NT + t) * NH + h) * NM + m) * NDK;
    out[base + col]      = (_Float16)(o0[v] * inv);
    out[base + 16 + col] = (_Float16)(o1[v] * inv);
  }
}

// ---------------- Attention across m=32 per (b,t,h) ----------------
__global__ __launch_bounds__(256) void attn_m_kernel(const _Float16* __restrict__ a1,
                                                     _Float16* __restrict__ out) {
  __shared__ _Float16 Xs[8][32][32];
  __shared__ _Float16 Pl[8][32][32];
  int wave = (blockIdx.x * blockDim.x + threadIdx.x) >> 5;
  int h = wave & 3;
  int t = (wave >> 2) & (NT - 1);
  int b = wave >> 11;
  int w = threadIdx.x >> 5;
  int lane = threadIdx.x & 31;
  const _Float16* X = a1 + ((((size_t)b * NT + t) * NH + h) * NM) * NDK;  // 32x32 tile

  // Stage the 2 KB X tile into LDS with CDNA5 async copies (ASYNCcnt-tracked).
  unsigned           lbase = (unsigned)(uintptr_t)(&Xs[w][0][0]);
  unsigned long long gbase = (unsigned long long)(uintptr_t)X;
#pragma unroll
  for (int c2 = 0; c2 < 4; ++c2) {
    unsigned           loff = lbase + (unsigned)(c2 * 512 + lane * 16);
    unsigned long long ga   = gbase + (unsigned long long)(c2 * 512 + lane * 16);
    asm volatile("global_load_async_to_lds_b128 %0, %1, off" :: "v"(loff), "v"(ga) : "memory");
  }
  asm volatile("s_wait_asynccnt 0x0" ::: "memory");
  const _Float16* Xl = &Xs[w][0][0];

  int col = lane & 15;
  int hf  = lane >> 4;
  const float scale = 0.17677669529663689f;   // 1/sqrt(32)
  v16h xa0 = load_rm(Xl, 32);
  v16h xa1 = load_rm(Xl + 16 * 32, 32);
  v8f s00 = {}, s01 = {}, s10 = {}, s11 = {};
  s00 = wmma_f16(xa0, xa0, s00);
  s01 = wmma_f16(xa0, xa1, s01);
  s10 = wmma_f16(xa1, xa0, s10);
  s11 = wmma_f16(xa1, xa1, s11);
  _Float16 (*P)[32] = Pl[w];
#pragma unroll
  for (int v = 0; v < 8; ++v) {
    int row = v + (hf << 3);
    {
      float x0 = s00[v] * scale, x1 = s01[v] * scale;
      float mx = rmax16(fmaxf(x0, x1));
      float p0 = __expf(x0 - mx), p1 = __expf(x1 - mx);
      float inv = 1.0f / rsum16(p0 + p1);
      P[row][col]      = (_Float16)(p0 * inv);
      P[row][16 + col] = (_Float16)(p1 * inv);
    }
    {
      float x0 = s10[v] * scale, x1 = s11[v] * scale;
      float mx = rmax16(fmaxf(x0, x1));
      float p0 = __expf(x0 - mx), p1 = __expf(x1 - mx);
      float inv = 1.0f / rsum16(p0 + p1);
      P[16 + row][col]      = (_Float16)(p0 * inv);
      P[16 + row][16 + col] = (_Float16)(p1 * inv);
    }
  }
  __syncthreads();
  v16h pA0 = load_rm(&P[0][0], 32);
  v16h pA1 = load_rm(&P[16][0], 32);
  v16h bx0 = load_cm(Xl, 32);        // column gathers now hit LDS, not global
  v16h bx1 = load_cm(Xl + 16, 32);
  v8f o00 = {}, o01 = {}, o10 = {}, o11 = {};
  o00 = wmma_f16(pA0, bx0, o00);
  o01 = wmma_f16(pA0, bx1, o01);
  o10 = wmma_f16(pA1, bx0, o10);
  o11 = wmma_f16(pA1, bx1, o11);
  // out layout: [b][m][t][h*32+dk]
  int m0 = hf << 3;
#pragma unroll
  for (int v = 0; v < 8; ++v) {
    int mA = m0 + v;
    int mB = 16 + m0 + v;
    size_t baseA = (((size_t)b * NM + mA) * NT + t) * ND + h * NDK;
    size_t baseB = (((size_t)b * NM + mB) * NT + t) * ND + h * NDK;
    out[baseA + col]      = (_Float16)o00[v];
    out[baseA + 16 + col] = (_Float16)o01[v];
    out[baseB + col]      = (_Float16)o10[v];
    out[baseB + 16 + col] = (_Float16)o11[v];
  }
}

// ---------------- fp32 -> f16 conversion ----------------
__global__ void cvt_f16_kernel(const float* __restrict__ in, _Float16* __restrict__ out, int n) {
  int i = blockIdx.x * blockDim.x + threadIdx.x;
  if (i < n) out[i] = (_Float16)in[i];
}

extern "C" void kernel_launch(void* const* d_in, const int* in_sizes, int n_in,
                              void* d_out, int out_size, void* d_ws, size_t ws_size,
                              hipStream_t stream) {
  (void)in_sizes; (void)n_in; (void)out_size; (void)ws_size;
  const float* x   = (const float*)d_in[0];
  const float* Wq  = (const float*)d_in[1];
  const float* bq  = (const float*)d_in[2];
  const float* Wk  = (const float*)d_in[3];
  const float* bk  = (const float*)d_in[4];
  const float* Wv  = (const float*)d_in[5];
  const float* bv  = (const float*)d_in[6];
  const float* Wfc = (const float*)d_in[7];
  const float* bfc = (const float*)d_in[8];
  const float* W1  = (const float*)d_in[9];
  const float* b1  = (const float*)d_in[10];
  const float* W2  = (const float*)d_in[11];
  const float* b2  = (const float*)d_in[12];
  const float* g1  = (const float*)d_in[13];
  const float* be1 = (const float*)d_in[14];
  const float* g2  = (const float*)d_in[15];
  const float* be2 = (const float*)d_in[16];
  const float* gf  = (const float*)d_in[17];
  const float* bef = (const float*)d_in[18];

  char* ws = (char*)d_ws;
  size_t off = 0;
  auto bump = [&](size_t bytes) -> char* {
    char* p = ws + off;
    off = (off + bytes + 255) & ~(size_t)255;
    return p;
  };
  float*    xw   = (float*)bump((size_t)NROW * ND * 4);
  _Float16* yh   = (_Float16*)bump((size_t)NROW * ND * 2);
  _Float16* qh   = (_Float16*)bump((size_t)NROW * ND * 2);
  _Float16* kh   = (_Float16*)bump((size_t)NROW * ND * 2);
  _Float16* qT   = (_Float16*)bump((size_t)NROW * ND * 2);
  _Float16* kT   = (_Float16*)bump((size_t)NROW * ND * 2);
  _Float16* vT   = (_Float16*)bump((size_t)NROW * ND * 2);
  _Float16* cqh  = (_Float16*)bump((size_t)NROW * ND * 2);
  _Float16* ckh  = (_Float16*)bump((size_t)NROW * ND * 2);
  _Float16* a1h  = (_Float16*)bump((size_t)NROW * ND * 2);
  _Float16* a2h  = (_Float16*)bump((size_t)NROW * ND * 2);
  _Float16* hid  = (_Float16*)bump((size_t)NROW * NFF * 2);
  _Float16* wqh  = (_Float16*)bump((size_t)2 * ND * ND * 2);
  _Float16* wkh  = (_Float16*)bump((size_t)2 * ND * ND * 2);
  _Float16* wvh  = (_Float16*)bump((size_t)2 * ND * ND * 2);
  _Float16* wfch = (_Float16*)bump((size_t)2 * ND * ND * 2);
  _Float16* w1h  = (_Float16*)bump((size_t)2 * NFF * ND * 2);
  _Float16* w2h  = (_Float16*)bump((size_t)2 * ND * NFF * 2);

  hipMemcpyAsync(xw, x, (size_t)NROW * ND * 4, hipMemcpyDeviceToDevice, stream);

  auto cvt = [&](const float* src, _Float16* dst, int n) {
    cvt_f16_kernel<<<(n + 255) / 256, 256, 0, stream>>>(src, dst, n);
  };
  cvt(Wq,  wqh,  2 * ND * ND);
  cvt(Wk,  wkh,  2 * ND * ND);
  cvt(Wv,  wvh,  2 * ND * ND);
  cvt(Wfc, wfch, 2 * ND * ND);
  cvt(W1,  w1h,  2 * NFF * ND);
  cvt(W2,  w2h,  2 * ND * NFF);

  dim3 blk(256);
  const int lnBlocks    = NROW / 8;                          // 4096
  const int gemmBlk128  = (NROW / 16) * (ND / 64) / 8;       // 512  (DO=128, 64-wide strips)
  const int gemmBlkFF   = (NROW / 16) * (NFF / 64) / 8;      // 2048 (DO=512)
  const int ctxBlocks   = (NB * NM * (NT / 16)) / 8;         // 256
  const int atBlocks    = (NB * NM * NH * (NT / 16)) / 8;    // 1024
  const int amBlocks    = (NB * NT * NH) / 8;                // 512

  for (int i = 0; i < 2; ++i) {
    ln_kernel<true><<<lnBlocks, blk, 0, stream>>>(xw, g1 + i * ND, be1 + i * ND, yh, nullptr);
    // q,k: row-major + transposed; v: transposed only
    gemm_kernel<ND, false, true, false, true><<<gemmBlk128, blk, 0, stream>>>(
        yh, wqh + (size_t)i * ND * ND, bq + i * ND, qh, qT, nullptr, nullptr, ND);
    gemm_kernel<ND, false, true, false, true><<<gemmBlk128, blk, 0, stream>>>(
        yh, wkh + (size_t)i * ND * ND, bk + i * ND, kh, kT, nullptr, nullptr, ND);
    gemm_kernel<ND, false, false, false, true><<<gemmBlk128, blk, 0, stream>>>(
        yh, wvh + (size_t)i * ND * ND, bv + i * ND, nullptr, vT, nullptr, nullptr, ND);
    ctx_attn_kernel<<<ctxBlocks, blk, 0, stream>>>(qh, qT, cqh);
    ctx_attn_kernel<<<ctxBlocks, blk, 0, stream>>>(kh, kT, ckh);
    attn_t_kernel<<<atBlocks, blk, 0, stream>>>(cqh, ckh, vT, a1h);
    attn_m_kernel<<<amBlocks, blk, 0, stream>>>(a1h, a2h);
    gemm_kernel<ND, false, false, true, false><<<gemmBlk128, blk, 0, stream>>>(
        a2h, wfch + (size_t)i * ND * ND, bfc + i * ND, nullptr, nullptr, xw, xw, ND);
    ln_kernel<true><<<lnBlocks, blk, 0, stream>>>(xw, g2 + i * ND, be2 + i * ND, yh, nullptr);
    gemm_kernel<ND, true, true, false, false><<<gemmBlkFF, blk, 0, stream>>>(
        yh, w1h + (size_t)i * NFF * ND, b1 + i * NFF, hid, nullptr, nullptr, nullptr, NFF);
    gemm_kernel<NFF, false, false, true, false><<<gemmBlk128, blk, 0, stream>>>(
        hid, w2h + (size_t)i * ND * NFF, b2 + i * ND, nullptr, nullptr, xw, xw, ND);
  }
  ln_kernel<false><<<lnBlocks, blk, 0, stream>>>(xw, gf, bef, nullptr, (float*)d_out);
}
